// TaGSim_78486232367693
// MI455X (gfx1250) — matrix operand order repaired and checked by reading, not compiled, not run on hardware
//
#include <hip/hip_runtime.h>
#include <math.h>

typedef __attribute__((ext_vector_type(16))) _Float16 v16h;
typedef __attribute__((ext_vector_type(8)))  float    v8f;

#define NN 8000
#define LL 32

// ---------------------------------------------------------------------------
// Transpose + convert: x[NN][32] f32 (row-major) -> xT[32][NN] f16
// ---------------------------------------------------------------------------
__global__ void k_transpose(const float* __restrict__ x, _Float16* __restrict__ xT) {
    int e = blockIdx.x * blockDim.x + threadIdx.x;
    if (e < NN * LL) {
        int row = e >> 5;
        int col = e & 31;
        xT[(size_t)col * NN + row] = (_Float16)x[e];
    }
}

// ---------------------------------------------------------------------------
// Column sums: out[32] = sum over rows of x[row][col]
// ---------------------------------------------------------------------------
__global__ void k_colsum(const float* __restrict__ x, float* __restrict__ out) {
    __shared__ float lds[32][32];
    int tid = threadIdx.x;           // 1024 threads
    int col = tid & 31;
    int r0  = tid >> 5;              // 0..31
    float p = 0.f;
    for (int row = r0; row < NN; row += 32) p += x[row * LL + col];
    lds[r0][col] = p;
    __syncthreads();
    if (tid < 32) {
        float s = 0.f;
        for (int r = 0; r < 32; ++r) s += lds[r][tid];
        out[tid] = s;
    }
}

// ---------------------------------------------------------------------------
// y = adj @ x  for both graphs in one launch.
// Block (256 thr = 8 waves) handles one 16-row x 32-col tile of y.
// Waves split K=8000 into 250 chunks of 32 (round-robin), WMMA f16->f32,
// then LDS-reduce the 8 wave partials.
// ---------------------------------------------------------------------------
__global__ void __launch_bounds__(256)
k_gemm(const float* __restrict__ adj1, const float* __restrict__ adj2,
       const _Float16* __restrict__ xT1, const _Float16* __restrict__ xT2,
       float* __restrict__ y1, float* __restrict__ y2) {
    __shared__ float red[8][16][32];   // 16 KB wave partials

    int bid = blockIdx.x;
    const float* adj; const _Float16* xT; float* y; int tile;
    if (bid < 500) { adj = adj1; xT = xT1; y = y1; tile = bid; }
    else           { adj = adj2; xT = xT2; y = y2; tile = bid - 500; }

    const int lane = threadIdx.x & 31;
    const int wv   = threadIdx.x >> 5;
    const int m    = lane & 15;        // sub-lane index
    const int hi   = lane >> 4;        // 0 for lanes 0-15, 1 for 16-31

    const int row = tile * 16 + m;
    const float*    arow  = adj + (size_t)row * NN;
    const int       abase = hi * 16;           // A halves 0..15 = K+abase..+15
    const _Float16* b0p   = xT + (size_t)m * NN;        // column m
    const _Float16* b1p   = xT + (size_t)(m + 16) * NN; // column m+16
    const int       kofsB = hi * 16;           // B lanes 16-31 hold K=16..31

    v8f c0 = {};
    v8f c1 = {};

    for (int c = wv; c < 250; c += 8) {
        const int k0 = c * 32;
        const float* ap = arow + k0 + abase;
        // 16-bit A 16x32, contiguous k order used here (B matches): halves 0..15 -> K=k0+abase+0..15
        float4 f0 = *(const float4*)(ap);
        float4 f1 = *(const float4*)(ap + 4);
        float4 f2 = *(const float4*)(ap + 8);
        float4 f3 = *(const float4*)(ap + 12);
        v16h a;
        a[0]  = (_Float16)f0.x; a[1]  = (_Float16)f0.y;
        a[2]  = (_Float16)f0.z; a[3]  = (_Float16)f0.w;
        a[4]  = (_Float16)f1.x; a[5]  = (_Float16)f1.y;
        a[6]  = (_Float16)f1.z; a[7]  = (_Float16)f1.w;
        a[8]  = (_Float16)f2.x; a[9]  = (_Float16)f2.y;
        a[10] = (_Float16)f2.z; a[11] = (_Float16)f2.w;
        a[12] = (_Float16)f3.x; a[13] = (_Float16)f3.y;
        a[14] = (_Float16)f3.z; a[15] = (_Float16)f3.w;

        // 16-bit B 32x16, same contiguous k order as A: lane holds 16 consecutive K for its column
        v16h b0 = *(const v16h*)(b0p + k0 + kofsB);
        v16h b1 = *(const v16h*)(b1p + k0 + kofsB);

        c0 = __builtin_amdgcn_wmma_f32_16x16x32_f16(false, a, false, b0,
                                                    (short)0, c0, false, false);
        c1 = __builtin_amdgcn_wmma_f32_16x16x32_f16(false, a, false, b1,
                                                    (short)0, c1, false, false);
    }

    // C/D layout: lanes 0-15: M=r, N=lane ; lanes 16-31: M=r+8, N=lane-16
    const int mbase = hi * 8;
    for (int r = 0; r < 8; ++r) {
        red[wv][mbase + r][m]      = c0[r];
        red[wv][mbase + r][16 + m] = c1[r];
    }
    __syncthreads();

    for (int e = threadIdx.x; e < 512; e += 256) {
        int mm = e >> 5;
        int nn = e & 31;
        float s = 0.f;
        for (int w = 0; w < 8; ++w) s += red[w][mm][nn];
        y[(size_t)(tile * 16 + mm) * LL + nn] = s;
    }
}

// ---------------------------------------------------------------------------
// Head: 3 branches (one wave each). g = [g1_01,g1_h1,g1_h2,g2_01,g2_h1,g2_h2]
// ---------------------------------------------------------------------------
__device__ __forceinline__ float frelu(float v) { return v > 0.f ? v : 0.f; }

__global__ void k_head(const float* __restrict__ g,
                       const float* __restrict__ tn_w,  const float* __restrict__ tn_wb,
                       const float* __restrict__ tn_bias,
                       const float* __restrict__ f1w, const float* __restrict__ f1b,
                       const float* __restrict__ f2w, const float* __restrict__ f2b,
                       const float* __restrict__ f3w, const float* __restrict__ f3b,
                       const float* __restrict__ scw, const float* __restrict__ scb,
                       float* __restrict__ out) {
    __shared__ float x1s[3][64], x2s[3][64];
    __shared__ float s1[3][16], s2[3][16], s3[3][8], s4[3][4];
    const int tid = threadIdx.x;    // 96 threads = 3 waves
    const int b = tid >> 5;         // branch
    const int k = tid & 31;

    // branches 0,1 use (g1_01|g1_h1) vs (g2_01|g2_h1); branch 2 uses h1|h2
    const int lo1 = (b == 2) ? 1 : 0;
    const int lo2 = (b == 2) ? 4 : 3;
    x1s[b][k]      = g[lo1 * 32 + k];
    x1s[b][k + 32] = g[(lo1 + 1) * 32 + k];
    x2s[b][k]      = g[lo2 * 32 + k];
    x2s[b][k + 32] = g[(lo2 + 1) * 32 + k];
    __syncthreads();

    if (k < 16) {
        const float* w = tn_w + (size_t)b * 64 * 64 * 16;
        float bil = 0.f;
        for (int d = 0; d < 64; ++d) {
            float acc = 0.f;
            for (int e = 0; e < 64; ++e) acc += w[d * 1024 + e * 16 + k] * x2s[b][e];
            bil += x1s[b][d] * acc;
        }
        const float* wb = tn_wb + (size_t)b * 16 * 128 + k * 128;
        float blk = 0.f;
        for (int j = 0; j < 64; ++j) blk += wb[j] * x1s[b][j];
        for (int j = 0; j < 64; ++j) blk += wb[64 + j] * x2s[b][j];
        s1[b][k] = frelu(bil + blk + tn_bias[b * 16 + k]);
    }
    __syncthreads();
    if (k < 16) {
        const float* w = f1w + b * 16 * 16 + k * 16;
        float acc = f1b[b * 16 + k];
        for (int j = 0; j < 16; ++j) acc += w[j] * s1[b][j];
        s2[b][k] = frelu(acc);
    }
    __syncthreads();
    if (k < 8) {
        const float* w = f2w + b * 8 * 16 + k * 16;
        float acc = f2b[b * 8 + k];
        for (int j = 0; j < 16; ++j) acc += w[j] * s2[b][j];
        s3[b][k] = frelu(acc);
    }
    __syncthreads();
    if (k < 4) {
        const float* w = f3w + b * 4 * 8 + k * 8;
        float acc = f3b[b * 4 + k];
        for (int j = 0; j < 8; ++j) acc += w[j] * s3[b][j];
        s4[b][k] = frelu(acc);
    }
    __syncthreads();
    if (k == 0) {
        float acc = scb[b];
        for (int j = 0; j < 4; ++j) acc += scw[b * 4 + j] * s4[b][j];
        out[b] = 1.f / (1.f + expf(-acc));
    }
}

// ---------------------------------------------------------------------------
extern "C" void kernel_launch(void* const* d_in, const int* in_sizes, int n_in,
                              void* d_out, int out_size, void* d_ws, size_t ws_size,
                              hipStream_t stream) {
    const float* adj1   = (const float*)d_in[0];
    const float* adj2   = (const float*)d_in[1];
    const float* feat1  = (const float*)d_in[2];
    const float* feat2  = (const float*)d_in[3];
    const float* tn_w   = (const float*)d_in[4];
    const float* tn_wb  = (const float*)d_in[5];
    const float* tn_b   = (const float*)d_in[6];
    const float* f1w    = (const float*)d_in[7];
    const float* f1b    = (const float*)d_in[8];
    const float* f2w    = (const float*)d_in[9];
    const float* f2b    = (const float*)d_in[10];
    const float* f3w    = (const float*)d_in[11];
    const float* f3b    = (const float*)d_in[12];
    const float* scw    = (const float*)d_in[13];
    const float* scb    = (const float*)d_in[14];
    float* out = (float*)d_out;

    char* ws = (char*)d_ws;
    _Float16* xT1 = (_Float16*)(ws);               // 32*8000*2 = 512000 B
    _Float16* xT2 = (_Float16*)(ws + 512000);
    float* h11 = (float*)(ws + 1024000);           // 8000*32*4 = 1024000 B each
    float* h21 = (float*)(ws + 2048000);
    float* h12 = (float*)(ws + 3072000);
    float* h22 = (float*)(ws + 4096000);
    float* g   = (float*)(ws + 5120000);           // 6 x 32 floats

    const int tgrid = (NN * LL + 255) / 256;       // 1000

    // pass 1: prep inputs
    k_transpose<<<tgrid, 256, 0, stream>>>(feat1, xT1);
    k_transpose<<<tgrid, 256, 0, stream>>>(feat2, xT2);
    k_colsum<<<1, 1024, 0, stream>>>(feat1, g + 0);    // g1_01
    k_colsum<<<1, 1024, 0, stream>>>(feat2, g + 96);   // g2_01

    // h11 = adj1@feat1 ; h21 = adj2@feat2
    k_gemm<<<1000, 256, 0, stream>>>(adj1, adj2, xT1, xT2, h11, h21);

    // pass 2: prep h1 operands
    k_transpose<<<tgrid, 256, 0, stream>>>(h11, xT1);
    k_transpose<<<tgrid, 256, 0, stream>>>(h21, xT2);
    k_colsum<<<1, 1024, 0, stream>>>(h11, g + 32);     // g1_h1
    k_colsum<<<1, 1024, 0, stream>>>(h21, g + 128);    // g2_h1

    // h12 = adj1@h11 ; h22 = adj2@h21
    k_gemm<<<1000, 256, 0, stream>>>(adj1, adj2, xT1, xT2, h12, h22);

    k_colsum<<<1, 1024, 0, stream>>>(h12, g + 64);     // g1_h2
    k_colsum<<<1, 1024, 0, stream>>>(h22, g + 160);    // g2_h2

    // tiny 3-branch head -> out[3]
    k_head<<<1, 96, 0, stream>>>(g, tn_w, tn_wb, tn_b,
                                 f1w, f1b, f2w, f2b, f3w, f3b, scw, scb, out);
}
